// GINModel_53463752900650
// MI455X (gfx1250) — matrix operand order repaired and checked by reading, not compiled, not run on hardware
//
#include <hip/hip_runtime.h>
#include <hip/hip_bf16.h>

#define N_NODES 50000
#define N_EDGES 800000
#define CH 256
#define BN_EPS 1e-5f

typedef __attribute__((ext_vector_type(16))) _Float16 v16h;
typedef __attribute__((ext_vector_type(8)))  float    v8f;

// ---------------- prep: fold Linear bias + BatchNorm(eval) into (s, t) --------
__global__ void k_prep(const float* __restrict__ b1, const float* __restrict__ g1,
                       const float* __restrict__ be1, const float* __restrict__ m1,
                       const float* __restrict__ v1,
                       const float* __restrict__ b2, const float* __restrict__ g2,
                       const float* __restrict__ be2, const float* __restrict__ m2,
                       const float* __restrict__ v2,
                       float* __restrict__ s1, float* __restrict__ t1,
                       float* __restrict__ s2, float* __restrict__ t2) {
    int c = threadIdx.x;
    if (c < CH) {
        float sc1 = g1[c] * rsqrtf(v1[c] + BN_EPS);
        s1[c] = sc1;
        t1[c] = (b1[c] - m1[c]) * sc1 + be1[c];
        float sc2 = g2[c] * rsqrtf(v2[c] + BN_EPS);
        s2[c] = sc2;
        t2[c] = (b2[c] - m2[c]) * sc2 + be2[c];
    }
}

// ------------- weight swizzle into WMMA B-fragment layout, f16 hi/lo ----------
// W is f32 [K=256][N=256] row-major. Fragment array layout:
// idx = (((kt*16 + ntile)*32 + lane)*16 + e), lane = n + 16*g,
// element e of the v16h = W[kt*32 + 8*g + (e&7) + 16*(e>>3)][ntile*16 + n]  (ISA 16-bit B layout)
__global__ void k_wfrag(const float* __restrict__ W,
                        _Float16* __restrict__ whi, _Float16* __restrict__ wlo) {
    int idx = blockIdx.x * blockDim.x + threadIdx.x;
    if (idx >= 8 * 16 * 32 * 16) return;
    int e    = idx & 15;
    int lane = (idx >> 4) & 31;
    int nt   = (idx >> 9) & 15;
    int kt   = idx >> 13;
    int n = lane & 15, g = lane >> 4;
    int k   = kt * 32 + g * 8 + (e & 7) + 16 * (e >> 3);
    int col = nt * 16 + n;
    float w = W[k * CH + col];
    _Float16 h = (_Float16)w;
    whi[idx] = h;
    wlo[idx] = (_Float16)(w - (float)h);
}

// ------------------------------ CSR build ------------------------------------
__global__ void k_zero_i32(int* __restrict__ p, int n) {
    int i = blockIdx.x * blockDim.x + threadIdx.x;
    if (i < n) p[i] = 0;
}

__global__ void k_deg(const long long* __restrict__ ei, int* __restrict__ deg) {
    int e = blockIdx.x * blockDim.x + threadIdx.x;
    if (e < N_EDGES) {
        int dst = (int)ei[N_EDGES + e];
        atomicAdd(&deg[dst], 1);
    }
}

// single-block 1024-thread chunked inclusive scan -> rowptr[0..N_NODES]
__global__ __launch_bounds__(1024) void k_scan(const int* __restrict__ deg,
                                               int* __restrict__ rowptr) {
    __shared__ int wsum[32];
    int tid  = threadIdx.x;
    int lane = tid & 31, wid = tid >> 5;
    if (tid == 0) rowptr[0] = 0;
    int running = 0;
    for (int base = 0; base < N_NODES; base += 1024) {
        int i = base + tid;
        int v = (i < N_NODES) ? deg[i] : 0;
        int x = v;
#pragma unroll
        for (int off = 1; off < 32; off <<= 1) {
            int y = __shfl_up(x, off);
            if (lane >= off) x += y;
        }
        if (lane == 31) wsum[wid] = x;
        __syncthreads();
        if (wid == 0) {
            int s = wsum[lane];
#pragma unroll
            for (int off = 1; off < 32; off <<= 1) {
                int y = __shfl_up(s, off);
                if (lane >= off) s += y;
            }
            wsum[lane] = s;
        }
        __syncthreads();
        int incl = x + (wid ? wsum[wid - 1] : 0) + running;
        if (i < N_NODES) rowptr[i + 1] = incl;
        running += wsum[31];
        __syncthreads();   // before wsum is reused next chunk
    }
}

__global__ void k_cursor(const int* __restrict__ rowptr, int* __restrict__ cursor) {
    int i = blockIdx.x * blockDim.x + threadIdx.x;
    if (i < N_NODES) cursor[i] = rowptr[i];
}

__global__ void k_fill(const long long* __restrict__ ei,
                       int* __restrict__ cursor, int* __restrict__ col) {
    int e = blockIdx.x * blockDim.x + threadIdx.x;
    if (e < N_EDGES) {
        int src = (int)ei[e];
        int dst = (int)ei[N_EDGES + e];
        int p = atomicAdd(&cursor[dst], 1);
        col[p] = src;
    }
}

// --------------- gather-sum aggregation: out[i] = x[i] + sum_j x[col[j]] -----
// one wave (32 lanes) per node; lane covers 8 channels -> 1KB coalesced rows
__global__ __launch_bounds__(256) void k_gather(const float* __restrict__ xin,
                                                const int* __restrict__ rowptr,
                                                const int* __restrict__ col,
                                                float* __restrict__ out) {
    int lane = threadIdx.x & 31;
    int node = blockIdx.x * 8 + (threadIdx.x >> 5);
    if (node >= N_NODES) return;
    int c0 = lane * 8;
    const float4* own = (const float4*)(xin + (size_t)node * CH + c0);
    float4 a0 = own[0], a1 = own[1];
    int beg = rowptr[node], end = rowptr[node + 1];
    for (int j = beg; j < end; ++j) {
        int s = col[j];
        const float4* r = (const float4*)(xin + (size_t)s * CH + c0);
        float4 b0 = r[0], b1 = r[1];
        a0.x += b0.x; a0.y += b0.y; a0.z += b0.z; a0.w += b0.w;
        a1.x += b1.x; a1.y += b1.y; a1.z += b1.z; a1.w += b1.w;
    }
    float4* o = (float4*)(out + (size_t)node * CH + c0);
    o[0] = a0; o[1] = a1;
}

// ----------- WMMA GEMM + fused BN/ReLU:  out = relu((A@W)*s + t) -------------
// block = 256 thr = 8 waves; block tile 64(M) x 256(N); wave tile 16 x 128.
// fp32 accuracy via f16 split: A*W ~= Ahi*Whi + Ahi*Wlo + Alo*Whi
__global__ __launch_bounds__(256) void k_gemm(const float* __restrict__ A,
                                              const _Float16* __restrict__ Whi,
                                              const _Float16* __restrict__ Wlo,
                                              const float* __restrict__ s,
                                              const float* __restrict__ t,
                                              float* __restrict__ out) {
    const int wave = threadIdx.x >> 5;    // 0..7
    const int lane = threadIdx.x & 31;
    const int wm = wave >> 1;             // 0..3 : M sub-tile
    const int wn = wave & 1;              // 0..1 : N half
    const int sub = lane & 15, g = lane >> 4;
    const int m0 = blockIdx.x * 64 + wm * 16;

    int row = m0 + sub;
    if (row >= N_NODES) row = N_NODES - 1;           // clamp loads, guard stores
    const float* arow = A + (size_t)row * CH;

    v8f acc[8];
#pragma unroll
    for (int i = 0; i < 8; ++i) { v8f z = {}; acc[i] = z; }

#pragma unroll
    for (int kt = 0; kt < 8; ++kt) {
        const int k0 = kt * 32;
        if (kt < 7) __builtin_prefetch(arow + k0 + 32, 0, 3);   // global_prefetch_b8
        // A fragment (ISA 16-bit A layout): lane (m+16g):
        //   elems 0..7  = K k0+8g+j ; elems 8..15 = K k0+16+8g+j
        const float4* p1 = (const float4*)(arow + k0 + 8 * g);
        const float4* p2 = (const float4*)(arow + k0 + 16 + 8 * g);
        float4 r1a = p1[0], r1b = p1[1];
        float4 r2a = p2[0], r2b = p2[1];
        float f1[8] = {r1a.x, r1a.y, r1a.z, r1a.w, r1b.x, r1b.y, r1b.z, r1b.w};
        float f2[8] = {r2a.x, r2a.y, r2a.z, r2a.w, r2b.x, r2b.y, r2b.z, r2b.w};
        v16h ah, al;
#pragma unroll
        for (int j = 0; j < 8; ++j) {
            _Float16 h1 = (_Float16)f1[j];
            ah[j] = h1;
            al[j] = (_Float16)(f1[j] - (float)h1);
            _Float16 h2 = (_Float16)f2[j];
            ah[8 + j] = h2;
            al[8 + j] = (_Float16)(f2[j] - (float)h2);
        }
#pragma unroll
        for (int nt = 0; nt < 8; ++nt) {
            int ntile = wn * 8 + nt;
            size_t fidx = (((size_t)(kt * 16 + ntile) * 32) + lane) * 16;
            v16h bh = *(const v16h*)(Whi + fidx);
            v16h bl = *(const v16h*)(Wlo + fidx);
            acc[nt] = __builtin_amdgcn_wmma_f32_16x16x32_f16(
                false, ah, false, bh, (short)0, acc[nt], false, false);
            acc[nt] = __builtin_amdgcn_wmma_f32_16x16x32_f16(
                false, ah, false, bl, (short)0, acc[nt], false, false);
            acc[nt] = __builtin_amdgcn_wmma_f32_16x16x32_f16(
                false, al, false, bh, (short)0, acc[nt], false, false);
        }
    }
    // epilogue: C/D layout lane (n+16g), elem r -> row m0+r+8g, col n
#pragma unroll
    for (int nt = 0; nt < 8; ++nt) {
        int c = wn * 128 + nt * 16 + sub;
        float sc = s[c], tc = t[c];
#pragma unroll
        for (int r = 0; r < 8; ++r) {
            int m = m0 + r + 8 * g;
            if (m < N_NODES) {
                float h = acc[nt][r] * sc + tc;
                out[(size_t)m * CH + c] = h > 0.f ? h : 0.f;
            }
        }
    }
}

// ------------------------- classifier: out = h @ Wc + bc ---------------------
__global__ __launch_bounds__(256) void k_cls(const float* __restrict__ h,
                                             const float* __restrict__ Wc,
                                             const float* __restrict__ bc,
                                             float* __restrict__ out) {
    int lane = threadIdx.x & 31;
    int node = blockIdx.x * 8 + (threadIdx.x >> 5);
    if (node >= N_NODES) return;
    int c0 = lane * 8;
    const float4* r = (const float4*)(h + (size_t)node * CH + c0);
    float4 a0 = r[0], a1 = r[1];
    float hv[8] = {a0.x, a0.y, a0.z, a0.w, a1.x, a1.y, a1.z, a1.w};
    float acc0 = 0.f, acc1 = 0.f;
#pragma unroll
    for (int j = 0; j < 8; ++j) {
        float2 w = ((const float2*)Wc)[c0 + j];   // Wc [256][2]
        acc0 += hv[j] * w.x;
        acc1 += hv[j] * w.y;
    }
    for (int off = 16; off > 0; off >>= 1) {
        acc0 += __shfl_down(acc0, off);
        acc1 += __shfl_down(acc1, off);
    }
    if (lane == 0) {
        out[(size_t)node * 2 + 0] = acc0 + bc[0];
        out[(size_t)node * 2 + 1] = acc1 + bc[1];
    }
}

// -----------------------------------------------------------------------------
extern "C" void kernel_launch(void* const* d_in, const int* in_sizes, int n_in,
                              void* d_out, int out_size, void* d_ws, size_t ws_size,
                              hipStream_t stream) {
    const float*     x   = (const float*)d_in[0];
    const long long* ei  = (const long long*)d_in[1];   // int64 [2][N_EDGES]
    const float* W1 = (const float*)d_in[2];
    const float* b1 = (const float*)d_in[3];
    const float* g1 = (const float*)d_in[4];
    const float* be1 = (const float*)d_in[5];
    const float* m1 = (const float*)d_in[6];
    const float* v1 = (const float*)d_in[7];
    const float* W2 = (const float*)d_in[8];
    const float* b2 = (const float*)d_in[9];
    const float* g2 = (const float*)d_in[10];
    const float* be2 = (const float*)d_in[11];
    const float* m2 = (const float*)d_in[12];
    const float* v2 = (const float*)d_in[13];
    const float* Wc = (const float*)d_in[14];
    const float* bc = (const float*)d_in[15];
    float* out = (float*)d_out;

    // workspace carve-up
    char* w = (char*)d_ws;
    float* buf0 = (float*)w; w += (size_t)N_NODES * CH * sizeof(float);
    float* buf1 = (float*)w; w += (size_t)N_NODES * CH * sizeof(float);
    const size_t FRAG = (size_t)8 * 16 * 32 * 16;     // 65536 halves = 128KB
    _Float16* w1hi = (_Float16*)w; w += FRAG * 2;
    _Float16* w1lo = (_Float16*)w; w += FRAG * 2;
    _Float16* w2hi = (_Float16*)w; w += FRAG * 2;
    _Float16* w2lo = (_Float16*)w; w += FRAG * 2;
    float* s1 = (float*)w; w += CH * sizeof(float);
    float* t1 = (float*)w; w += CH * sizeof(float);
    float* s2 = (float*)w; w += CH * sizeof(float);
    float* t2 = (float*)w; w += CH * sizeof(float);
    int* deg    = (int*)w; w += (size_t)N_NODES * sizeof(int);
    int* rowptr = (int*)w; w += (size_t)(N_NODES + 1) * sizeof(int);
    int* cursor = (int*)w; w += (size_t)N_NODES * sizeof(int);
    int* col    = (int*)w; w += (size_t)N_EDGES * sizeof(int);

    // parameter prep
    k_prep<<<1, 256, 0, stream>>>(b1, g1, be1, m1, v1, b2, g2, be2, m2, v2,
                                  s1, t1, s2, t2);
    k_wfrag<<<256, 256, 0, stream>>>(W1, w1hi, w1lo);
    k_wfrag<<<256, 256, 0, stream>>>(W2, w2hi, w2lo);

    // CSR build (once; reused by both layers)
    k_zero_i32<<<(N_NODES + 255) / 256, 256, 0, stream>>>(deg, N_NODES);
    k_deg<<<(N_EDGES + 255) / 256, 256, 0, stream>>>(ei, deg);
    k_scan<<<1, 1024, 0, stream>>>(deg, rowptr);
    k_cursor<<<(N_NODES + 255) / 256, 256, 0, stream>>>(rowptr, cursor);
    k_fill<<<(N_EDGES + 255) / 256, 256, 0, stream>>>(ei, cursor, col);

    const int GATHER_BLOCKS = (N_NODES + 7) / 8;      // 6250
    const int GEMM_BLOCKS   = (N_NODES + 63) / 64;    // 782

    // layer 1
    k_gather<<<GATHER_BLOCKS, 256, 0, stream>>>(x, rowptr, col, buf0);
    k_gemm<<<GEMM_BLOCKS, 256, 0, stream>>>(buf0, w1hi, w1lo, s1, t1, buf1);
    // layer 2
    k_gather<<<GATHER_BLOCKS, 256, 0, stream>>>(buf1, rowptr, col, buf0);
    k_gemm<<<GEMM_BLOCKS, 256, 0, stream>>>(buf0, w2hi, w2lo, s2, t2, buf1);
    // classifier
    k_cls<<<GATHER_BLOCKS, 256, 0, stream>>>(buf1, Wc, bc, out);
}
